// AttGRUModel_65274912964685
// MI455X (gfx1250) — matrix operand order repaired and checked by reading, compiled, not run on hardware
//
#include <hip/hip_runtime.h>
#include <math.h>

typedef __bf16 bf16_t;
typedef __attribute__((ext_vector_type(16))) __bf16 v16bf;
typedef __attribute__((ext_vector_type(8)))  float  v8f;

#define B_   32
#define S_   200
#define T_   49      // L-1
#define DE_  256
#define H_   512
#define G3_  1536    // 3H
#define VS_  30000

// ---------- helpers ----------
__device__ __forceinline__ bf16_t f2bf(float x) {
  unsigned u = __builtin_bit_cast(unsigned, x);
  unsigned r = (u + 0x7FFFu + ((u >> 16) & 1u)) >> 16;   // round-to-nearest-even
  unsigned short s = (unsigned short)r;
  return __builtin_bit_cast(bf16_t, s);
}
__device__ __forceinline__ float sigm(float x) { return 1.f / (1.f + __expf(-x)); }

// A fragment: row-major A[M x K], 16x32 tile at (m0,k0).  This code's contiguous k order (k = 16h + s):
// lanes 0-15 hold rows m, K 0..15 across v0-7; lanes 16-31 hold K 16..31 (same k order as B).
// Per lane this is one contiguous 32-byte chunk -> two global_load_b128.
__device__ __forceinline__ v16bf load_a16x32(const bf16_t* A, int lda, int m0, int k0, int lane) {
  const int m  = lane & 15;
  const int kb = (lane >> 4) << 4;
  const bf16_t* p = A + (size_t)(m0 + m) * lda + k0;
  v16bf f;
#pragma unroll
  for (int v = 0; v < 8; ++v) {
    const int kk = kb + (v << 1);
    f[2*v]   = p[kk];
    f[2*v+1] = p[kk+1];
  }
  return f;
}

// B fragment: weights stored [N x K] row-major (B[k][n] = W[n][k]).
// lanes 0-15: n=lane, K 0..15 across v0-7; lanes 16-31: n=lane-16, K 16..31.
// Per lane: one contiguous 32-byte chunk -> two global_load_b128.
__device__ __forceinline__ v16bf load_b32x16(const bf16_t* W, int ldw, int n0, int k0, int lane) {
  const int n  = lane & 15;
  const int kb = (lane >> 4) << 4;
  const bf16_t* p = W + (size_t)(n0 + n) * ldw + k0 + kb;
  v16bf f;
#pragma unroll
  for (int v = 0; v < 8; ++v) { f[2*v] = p[2*v]; f[2*v+1] = p[2*v+1]; }
  return f;
}

// grid-wide barrier (monotonic counter, zeroed per launch by k_init)
__device__ __forceinline__ void gsync(unsigned* bar, unsigned target) {
  __threadfence();
  __syncthreads();
  if (threadIdx.x == 0) {
    atomicAdd(bar, 1u);
    while (*(volatile unsigned*)bar < target) { __builtin_amdgcn_s_sleep(1); }
  }
  __syncthreads();
  __threadfence();
}

// ---------- tiny kernels ----------
__global__ void k_init(unsigned* bars) {
  if (blockIdx.x == 0 && threadIdx.x < 8) bars[threadIdx.x] = 0u;
}

__global__ void k_cvt_bf16(const float* __restrict__ in, bf16_t* __restrict__ out, int n) {
  for (int i = blockIdx.x * blockDim.x + threadIdx.x; i < n; i += gridDim.x * blockDim.x)
    out[i] = f2bf(in[i]);
}

// embedding gather: row r = t*32+b, token = tok[b*tokStride + t]
__global__ void k_gather(const int* __restrict__ tok, const float* __restrict__ emb,
                         bf16_t* __restrict__ out, int rows, int tokStride) {
  const int total = rows * DE_;
  for (int i = blockIdx.x * blockDim.x + threadIdx.x; i < total; i += gridDim.x * blockDim.x) {
    const int r = i / DE_, d = i - r * DE_;
    const int t = r >> 5, b = r & 31;
    const int tk = tok[b * tokStride + t];
    out[i] = f2bf(emb[(size_t)tk * DE_ + d]);
  }
}

// ---------- generic WMMA GEMM: C[M,N] = A[M,K] * W[N,K]^T + bias ----------
// Each wave computes a 16 x (16*NV) tile; A-fragment reused across the NV B-tiles.
// MODE 0: outF fp32;  MODE 1: outB = bf16(relu(.));  MODE 2: outF, row (t*32+b) -> out[(b*49+t)*N+col]
template <int MODE, int NV>
__global__ void k_wmma_gemm(const bf16_t* __restrict__ A, const bf16_t* __restrict__ W,
                            const float* __restrict__ bias,
                            float* __restrict__ outF, bf16_t* __restrict__ outB,
                            int M, int N, int K, int njBase, int nChunks) {
  const int lane = threadIdx.x & 31;
  // wave index is uniform: force it scalar so all tile control flow stays on the SALU
  const int gw = __builtin_amdgcn_readfirstlane((blockIdx.x * blockDim.x + threadIdx.x) >> 5);
  const int mT = M >> 4;
  if (gw >= mT * nChunks) return;
  const int mi  = gw / nChunks;
  const int njc = gw - mi * nChunks;
  const int m0  = mi << 4;
  const int n0  = (njBase + njc) << 6;

  v8f acc[NV] = {};
  for (int k0 = 0; k0 < K; k0 += 32) {
    const v16bf a = load_a16x32(A, K, m0, k0, lane);
#pragma unroll
    for (int j = 0; j < NV; ++j) {
      const v16bf b = load_b32x16(W, K, n0 + (j << 4), k0, lane);
      acc[j] = __builtin_amdgcn_wmma_f32_16x16x32_bf16(false, a, false, b,
                                                       (short)0, acc[j], false, false);
    }
  }
  const int l8 = (lane >> 4) << 3;
  const int cn = lane & 15;
#pragma unroll
  for (int j = 0; j < NV; ++j) {
    const int col = n0 + (j << 4) + cn;
    const float bv = bias[col];
#pragma unroll
    for (int r = 0; r < 8; ++r) {
      const int row = m0 + r + l8;
      const float v = acc[j][r] + bv;
      if constexpr (MODE == 0) {
        outF[(size_t)row * N + col] = v;
      } else if constexpr (MODE == 1) {
        outB[(size_t)row * N + col] = f2bf(fmaxf(v, 0.f));
      } else {
        const int b = row & 31, t = row >> 5;
        outF[((size_t)b * T_ + t) * N + col] = v;
      }
    }
  }
}

// ---------- recurrence building blocks ----------
// GH[32 x 1536] = h_bf16[32 x 512] @ Whh_bf16[1536 x 512]^T + bhh   (192 waves, one 16x16 tile each)
__device__ __forceinline__ void gh_tile(const bf16_t* __restrict__ hb, const bf16_t* __restrict__ Whh,
                                        const float* __restrict__ bhh, float* __restrict__ GH,
                                        int wave, int lane) {
  const int mi = wave & 1, nj = wave >> 1;
  v8f acc = {};
  for (int k0 = 0; k0 < H_; k0 += 32) {
    const v16bf a = load_a16x32(hb, H_, mi << 4, k0, lane);
    const v16bf b = load_b32x16(Whh, H_, nj << 4, k0, lane);
    acc = __builtin_amdgcn_wmma_f32_16x16x32_bf16(false, a, false, b, (short)0, acc, false, false);
  }
  const int l8  = (lane >> 4) << 3;
  const int col = (nj << 4) + (lane & 15);
  const float bv = bhh[col];
#pragma unroll
  for (int r = 0; r < 8; ++r)
    GH[((mi << 4) + r + l8) * G3_ + col] = acc[r] + bv;
}

__device__ __forceinline__ void gru_update(const float* __restrict__ gx, const float* __restrict__ GH,
                                           float* __restrict__ h, bf16_t* __restrict__ hb,
                                           float* __restrict__ enc_out, int t,
                                           int gtid, int nth) {
  for (int i = gtid; i < B_ * H_; i += nth) {
    const int b = i >> 9, j = i & (H_ - 1);
    const float* gxb = gx + (size_t)b * G3_;
    const float* ghb = GH + (size_t)b * G3_;
    const float r_ = sigm(gxb[j] + ghb[j]);
    const float z_ = sigm(gxb[H_ + j] + ghb[H_ + j]);
    const float n_ = tanhf(gxb[2 * H_ + j] + r_ * ghb[2 * H_ + j]);
    const float hn = (1.f - z_) * n_ + z_ * h[i];
    h[i] = hn;
    hb[i] = f2bf(hn);
    if (enc_out) enc_out[((size_t)b * S_ + t) * H_ + j] = hn;
  }
}

// ---------- persistent encoder: 200 sequential GRU steps ----------
__global__ void k_enc_rnn(const bf16_t* __restrict__ Whh, const float* __restrict__ bhh,
                          const float* __restrict__ GX, float* __restrict__ h, bf16_t* __restrict__ hb,
                          float* __restrict__ GH, float* __restrict__ enc_out, unsigned* bar) {
  const int gtid = blockIdx.x * blockDim.x + threadIdx.x;
  const int nth  = gridDim.x * blockDim.x;
  const int wave = __builtin_amdgcn_readfirstlane(gtid >> 5);
  const int lane = threadIdx.x & 31;
  const unsigned nb = gridDim.x;
  unsigned ep = 0;
  for (int i = gtid; i < B_ * H_; i += nth) { h[i] = 0.f; hb[i] = f2bf(0.f); }
  gsync(bar, (++ep) * nb);
  for (int t = 0; t < S_; ++t) {
    if (wave < 192) gh_tile(hb, Whh, bhh, GH, wave, lane);
    gsync(bar, (++ep) * nb);
    gru_update(GX + (size_t)t * B_ * G3_, GH, h, hb, enc_out, t, gtid, nth);
    gsync(bar, (++ep) * nb);
  }
}

// ---------- persistent decoder: 49 sequential GRU+attention steps ----------
__global__ void k_dec_rnn(const bf16_t* __restrict__ Whh, const float* __restrict__ bhh,
                          const float* __restrict__ GX, float* __restrict__ h, bf16_t* __restrict__ hb,
                          float* __restrict__ GH, const float* __restrict__ enc_out,
                          bf16_t* __restrict__ Cbuf, unsigned* bar) {
  __shared__ float s_h[H_];
  __shared__ float s_sc[256];
  __shared__ float s_red[256];
  const int tid  = threadIdx.x;
  const int gtid = blockIdx.x * blockDim.x + tid;
  const int nth  = gridDim.x * blockDim.x;
  const int wave = __builtin_amdgcn_readfirstlane(gtid >> 5);
  const int lane = tid & 31;
  const int b    = blockIdx.x;                 // 32 blocks: block b owns batch b's attention
  const unsigned nb = gridDim.x;
  unsigned ep = 0;
  for (int t = 0; t < T_; ++t) {
    if (wave < 192) gh_tile(hb, Whh, bhh, GH, wave, lane);
    gsync(bar, (++ep) * nb);
    gru_update(GX + (size_t)t * B_ * G3_, GH, h, hb, nullptr, 0, gtid, nth);
    gsync(bar, (++ep) * nb);
    // ---- attention over enc_out[b, :, :] with current h[b] ----
    for (int j = tid; j < H_; j += 256) s_h[j] = h[b * H_ + j];
    __syncthreads();
    float sc = 0.f;
    if (tid < S_) {
      const float* e = enc_out + ((size_t)b * S_ + tid) * H_;
      for (int j = 0; j < H_; ++j) sc += s_h[j] * e[j];
    }
    s_sc[tid]  = sc;
    s_red[tid] = (tid < S_) ? sc : -3.4e38f;
    __syncthreads();
    for (int w = 128; w > 0; w >>= 1) { if (tid < w) s_red[tid] = fmaxf(s_red[tid], s_red[tid + w]); __syncthreads(); }
    const float mx = s_red[0];
    __syncthreads();
    const float e1 = (tid < S_) ? __expf(s_sc[tid] - mx) : 0.f;
    s_sc[tid] = e1;
    s_red[tid] = e1;
    __syncthreads();
    for (int w = 128; w > 0; w >>= 1) { if (tid < w) s_red[tid] += s_red[tid + w]; __syncthreads(); }
    const float inv = 1.f / s_red[0];
    __syncthreads();
    if (tid < S_) s_sc[tid] *= inv;
    __syncthreads();
    for (int j = tid; j < H_; j += 256) {
      float c = 0.f;
      for (int s = 0; s < S_; ++s) c += s_sc[s] * enc_out[((size_t)b * S_ + s) * H_ + j];
      bf16_t* crow = Cbuf + ((size_t)t * B_ + b) * (2 * H_);
      crow[j]      = f2bf(c);        // ctx first ...
      crow[H_ + j] = f2bf(s_h[j]);   // ... then h  (concat order of reference)
    }
    gsync(bar, (++ep) * nb);
  }
}

// ---------- host-side GEMM launcher (handles N % 64 tail with a static-NV tail grid) ----------
template <int MODE>
static void launch_gemm(const bf16_t* A, const bf16_t* W, const float* bias,
                        float* oF, bf16_t* oB, int M, int N, int K, hipStream_t stream) {
  const int mT = M / 16, nT = N / 16;
  const int nFull = nT / 4, nRem = nT - nFull * 4;
  if (nFull > 0) {
    const int waves = mT * nFull;
    k_wmma_gemm<MODE, 4><<<(waves + 7) / 8, 256, 0, stream>>>(A, W, bias, oF, oB, M, N, K, 0, nFull);
  }
  if (nRem == 3) {
    k_wmma_gemm<MODE, 3><<<(mT + 7) / 8, 256, 0, stream>>>(A, W, bias, oF, oB, M, N, K, nFull, 1);
  } else if (nRem == 2) {
    k_wmma_gemm<MODE, 2><<<(mT + 7) / 8, 256, 0, stream>>>(A, W, bias, oF, oB, M, N, K, nFull, 1);
  } else if (nRem == 1) {
    k_wmma_gemm<MODE, 1><<<(mT + 7) / 8, 256, 0, stream>>>(A, W, bias, oF, oB, M, N, K, nFull, 1);
  }
}

// ---------- host ----------
extern "C" void kernel_launch(void* const* d_in, const int* in_sizes, int n_in,
                              void* d_out, int out_size, void* d_ws, size_t ws_size,
                              hipStream_t stream) {
  (void)in_sizes; (void)n_in; (void)out_size; (void)ws_size;
  const int*   code   = (const int*)  d_in[0];
  const int*   summ   = (const int*)  d_in[1];
  const float* encEmb = (const float*)d_in[3];
  const float* encWih = (const float*)d_in[4];
  const float* encWhh = (const float*)d_in[5];
  const float* encBih = (const float*)d_in[6];
  const float* encBhh = (const float*)d_in[7];
  const float* decEmb = (const float*)d_in[8];
  const float* decWih = (const float*)d_in[9];
  const float* decWhh = (const float*)d_in[10];
  const float* decBih = (const float*)d_in[11];
  const float* decBhh = (const float*)d_in[12];
  const float* W1     = (const float*)d_in[13];
  const float* b1     = (const float*)d_in[14];
  const float* W2     = (const float*)d_in[15];
  const float* b2     = (const float*)d_in[16];
  float* out = (float*)d_out;

  char* ws = (char*)d_ws;
  size_t off = 0;
  auto alloc = [&](size_t bytes) -> char* {
    char* p = ws + off;
    off = (off + bytes + 255) & ~(size_t)255;
    return p;
  };
  unsigned* bars   = (unsigned*)alloc(256);
  bf16_t*   WihEb  = (bf16_t*)alloc((size_t)G3_ * DE_ * 2);
  bf16_t*   WhhEb  = (bf16_t*)alloc((size_t)G3_ * H_  * 2);
  bf16_t*   WihDb  = (bf16_t*)alloc((size_t)G3_ * DE_ * 2);
  bf16_t*   WhhDb  = (bf16_t*)alloc((size_t)G3_ * H_  * 2);
  bf16_t*   W1b    = (bf16_t*)alloc((size_t)H_  * 2 * H_ * 2);
  bf16_t*   W2b    = (bf16_t*)alloc((size_t)VS_ * H_ * 2);
  bf16_t*   Xe     = (bf16_t*)alloc((size_t)S_ * B_ * DE_ * 2);
  bf16_t*   Xd     = (bf16_t*)alloc((size_t)T_ * B_ * DE_ * 2);
  float*    GXe    = (float*) alloc((size_t)S_ * B_ * G3_ * 4);
  float*    GXd    = (float*) alloc((size_t)T_ * B_ * G3_ * 4);
  float*    encOut = (float*) alloc((size_t)B_ * S_ * H_ * 4);
  float*    GH     = (float*) alloc((size_t)B_ * G3_ * 4);
  float*    hF     = (float*) alloc((size_t)B_ * H_ * 4);
  bf16_t*   hB     = (bf16_t*)alloc((size_t)B_ * H_ * 2);
  bf16_t*   Cbuf   = (bf16_t*)alloc((size_t)T_ * B_ * 2 * H_ * 2);
  bf16_t*   Rbuf   = (bf16_t*)alloc((size_t)T_ * B_ * H_ * 2);

  k_init<<<1, 64, 0, stream>>>(bars);

  auto cvt = [&](const float* src, bf16_t* dst, int n) {
    int blocks = (n + 255) / 256; if (blocks > 4096) blocks = 4096;
    k_cvt_bf16<<<blocks, 256, 0, stream>>>(src, dst, n);
  };
  cvt(encWih, WihEb, G3_ * DE_);
  cvt(encWhh, WhhEb, G3_ * H_);
  cvt(decWih, WihDb, G3_ * DE_);
  cvt(decWhh, WhhDb, G3_ * H_);
  cvt(W1,     W1b,   H_ * 2 * H_);
  cvt(W2,     W2b,   VS_ * H_);

  { int n = S_ * B_ * DE_; int blocks = (n + 255) / 256; if (blocks > 4096) blocks = 4096;
    k_gather<<<blocks, 256, 0, stream>>>(code, encEmb, Xe, S_ * B_, S_); }
  { int n = T_ * B_ * DE_; int blocks = (n + 255) / 256; if (blocks > 4096) blocks = 4096;
    k_gather<<<blocks, 256, 0, stream>>>(summ, decEmb, Xd, T_ * B_, 50); }

  // batched input-side GRU GEMMs (gx = x @ Wih^T + bih)
  launch_gemm<0>(Xe, WihEb, encBih, GXe, nullptr, S_ * B_, G3_, DE_, stream);
  launch_gemm<0>(Xd, WihDb, decBih, GXd, nullptr, T_ * B_, G3_, DE_, stream);

  // sequential encoder recurrence (24 blocks * 8 waves == 192 GH tiles)
  k_enc_rnn<<<24, 256, 0, stream>>>(WhhEb, encBhh, GXe, hF, hB, GH, encOut, bars + 0);

  // sequential decoder recurrence + attention (32 blocks, block b = batch b)
  k_dec_rnn<<<32, 256, 0, stream>>>(WhhDb, decBhh, GXd, hF, hB, GH, encOut, Cbuf, bars + 1);

  // deferred output projection, batched over all (t,b)
  launch_gemm<1>(Cbuf, W1b, b1, nullptr, Rbuf, T_ * B_, H_, 2 * H_, stream);   // relu(c@W1^T+b1) -> bf16
  launch_gemm<2>(Rbuf, W2b, b2, out, nullptr, T_ * B_, VS_, H_, stream);       // @W2^T+b2, permuted into d_out
}
